// Model_20658792693861
// MI455X (gfx1250) — matrix operand order repaired and checked by reading, not compiled, not run on hardware
//
#include <hip/hip_runtime.h>

typedef _Float16 h8   __attribute__((ext_vector_type(8)));
typedef _Float16 v16h __attribute__((ext_vector_type(16)));
typedef float    v8f  __attribute__((ext_vector_type(8)));

// ---------------- model constants ----------------
constexpr int cB    = 64;
constexpr int cSEQ  = 2048;
constexpr int cCIN  = 32;
constexpr int cPN   = 128;   // patches
constexpr int cL    = 129;   // tokens (patches + global)
constexpr int cD    = 512;
constexpr int cH    = 8;
constexpr int cDK   = 64;
constexpr int cDFF  = 2048;
constexpr int cPRED = 96;
constexpr int cCS   = 35;    // cross source length (31 chans + 4 marks)
constexpr int cSP   = 160;   // padded self seq (129 -> 160, mult of 32)
constexpr int cCP   = 64;    // padded cross seq (35 -> 64)
constexpr int cSBINS = 65;   // rfft bins for S=129
constexpr int cCBINS = 18;   // rfft bins for S=35
constexpr float cSCALE = 0.125f; // 1/sqrt(64)

// One pipeline stage: A frag (16x32) + 4 B frags (32x16 each)
struct Frag { h8 a0, a1, b0[4], b1[4]; };

__device__ __forceinline__ void fload(Frag& f, const _Float16* __restrict__ Ar,
                                      const _Float16* const* Wr, int k)
{
    f.a0 = *(const h8*)(Ar + k);
    f.a1 = *(const h8*)(Ar + k + 16);
#pragma unroll
    for (int s = 0; s < 4; ++s) {
        f.b0[s] = *(const h8*)(Wr[s] + k);
        f.b1[s] = *(const h8*)(Wr[s] + k + 16);
    }
}

__device__ __forceinline__ void fmma(v8f* acc, const Frag& f)
{
    v16h av = __builtin_shufflevector(f.a0, f.a1, 0,1,2,3,4,5,6,7,8,9,10,11,12,13,14,15);
#pragma unroll
    for (int s = 0; s < 4; ++s) {
        v16h bv = __builtin_shufflevector(f.b0[s], f.b1[s],
                                          0,1,2,3,4,5,6,7,8,9,10,11,12,13,14,15);
        acc[s] = __builtin_amdgcn_wmma_f32_16x16x32_f16(false, av, false, bv, (short)0,
                                                        acc[s], false, false);
    }
}

// ---------------- generic WMMA GEMM: C = act((A @ W^T + bias) * gscale) ----
// A: (M,K) f16 row-major, W: (N,K) f16 row-major.
// Each wave computes a 16x64 C tile (4 accumulators, A-frag reused across 4
// v_wmma_f32_16x16x32_f16). Ping-pong pipelined K-loop (unroll-by-2, two fixed
// fragment sets -> no register rotation copies, partial loadcnt waits).
// flags: 1=relu, 2=f16 out, 4=transposed C store, 8=per-batch gscale.
__global__ __launch_bounds__(128) void k_gemm(
    const _Float16* __restrict__ A, int lda, long sAo, long sAi,
    const _Float16* __restrict__ W, int ldw, long sWo, long sWi,
    float* __restrict__ C, _Float16* __restrict__ C16, int ldc, long sCo, long sCi,
    const float* __restrict__ bias, const float* __restrict__ gscale,
    int M, int N, int K, int bdiv, int flags)
{
    int z  = blockIdx.z;
    int zo = z / bdiv, zi = z - zo * bdiv;
    int wave = threadIdx.x >> 5;
    int tn64 = blockIdx.y * 4 + wave;
    if (tn64 * 64 >= N) return;          // uniform per wave; EXEC stays all-ones
    int tm   = blockIdx.x;
    int lane = threadIdx.x & 31;
    int half = lane >> 4;                // 0: lanes 0-15, 1: lanes 16-31
    int rl   = lane & 15;
    int arow = tm * 16 + rl; if (arow >= M) arow = M - 1;   // clamp edge tiles
    const _Float16* Ar = A + zo * sAo + zi * sAi + (long)arow * lda + half * 8;
    const _Float16* Wb = W + zo * sWo + zi * sWi + (long)half * 8;
    const _Float16* Wr[4];
#pragma unroll
    for (int s = 0; s < 4; ++s) {
        int wrow = tn64 * 64 + s * 16 + rl; if (wrow >= N) wrow = N - 1;
        Wr[s] = Wb + (long)wrow * ldw;
    }
    v8f acc[4];
#pragma unroll
    for (int s = 0; s < 4; ++s) acc[s] = (v8f){0.f,0.f,0.f,0.f,0.f,0.f,0.f,0.f};

    // --- ping-pong pipelined K loop ---
    Frag f0, f1;
    fload(f0, Ar, Wr, 0);
    int k = 0;
    for (; k + 64 <= K; k += 64) {
        fload(f1, Ar, Wr, k + 32);
        __builtin_prefetch(Ar + k + 256, 0, 1);
        __builtin_prefetch(Wr[0] + k + 256, 0, 1);
        __builtin_prefetch(Wr[1] + k + 256, 0, 1);
        __builtin_prefetch(Wr[2] + k + 256, 0, 1);
        __builtin_prefetch(Wr[3] + k + 256, 0, 1);
        fmma(acc, f0);                       // consumes loads issued one stage earlier
        int k2 = (k + 64 < K) ? (k + 64) : k;  // clamp final reload in-bounds
        fload(f0, Ar, Wr, k2);
        fmma(acc, f1);
    }
    if (k < K) fmma(acc, f0);                // leftover 32-wide step (K % 64 == 32)

    float gs = (flags & 8) ? gscale[zo] : 1.0f;
    long cbase = zo * sCo + zi * sCi;
#pragma unroll
    for (int s = 0; s < 4; ++s) {
        int col = tn64 * 64 + s * 16 + rl;
#pragma unroll
        for (int v = 0; v < 8; v++) {
            int row = tm * 16 + half * 8 + v;   // C layout: vgpr v -> M = v (+8 upper lanes)
            if (row < M && col < N) {
                float val = acc[s][v];
                if (bias) val += bias[col];
                val *= gs;
                if (flags & 1) val = fmaxf(val, 0.f);
                long off = (flags & 4) ? (cbase + (long)col * ldc + row)
                                       : (cbase + (long)row * ldc + col);
                if (flags & 2) C16[off] = (_Float16)val; else C[off] = val;
            }
        }
    }
}

// ---------------- per-(b,c) mean / stdev over time ----------------
__global__ __launch_bounds__(256) void k_stats(const float* __restrict__ xe,
                                               float* __restrict__ meanb,
                                               float* __restrict__ stdb)
{
    __shared__ float sb[256];
    int bc = blockIdx.x; int b = bc >> 5; int c = bc & 31;
    float s = 0.f, s2 = 0.f;
    for (int t = threadIdx.x; t < cSEQ; t += 256) {
        float v = xe[((long)b * cSEQ + t) * cCIN + c];
        s += v; s2 += v * v;
    }
    sb[threadIdx.x] = s; __syncthreads();
    for (int st = 128; st > 0; st >>= 1) { if (threadIdx.x < st) sb[threadIdx.x] += sb[threadIdx.x + st]; __syncthreads(); }
    float ts = sb[0]; __syncthreads();
    sb[threadIdx.x] = s2; __syncthreads();
    for (int st = 128; st > 0; st >>= 1) { if (threadIdx.x < st) sb[threadIdx.x] += sb[threadIdx.x + st]; __syncthreads(); }
    if (threadIdx.x == 0) {
        float m = ts / (float)cSEQ;
        float var = sb[0] / (float)cSEQ - m * m;
        meanb[bc] = m;
        stdb[bc]  = sqrtf(var + 1e-5f);
    }
}

// ---------------- normalize + build f16 patch rows / cross-input rows ------
__global__ __launch_bounds__(256) void k_prep(const float* __restrict__ xe,
                                              const float* __restrict__ xm,
                                              const float* __restrict__ meanb,
                                              const float* __restrict__ stdb,
                                              _Float16* __restrict__ xt16,   // (B,128,32) K-padded
                                              _Float16* __restrict__ ex16)   // (B,35,2048)
{
    int i = blockIdx.x * 256 + threadIdx.x;
    if (i >= cB * cSEQ) return;
    int b = i / cSEQ, t = i - b * cSEQ;
    long ebase = (long)b * cCS * cSEQ;
    for (int c = 0; c < cCIN; ++c) {
        float xn = (xe[((long)b * cSEQ + t) * cCIN + c] - meanb[b * cCIN + c]) / stdb[b * cCIN + c];
        if (c < cCIN - 1) ex16[ebase + (long)c * cSEQ + t] = (_Float16)xn;
        else {
            int p = t >> 4, j = t & 15;
            xt16[((long)b * cPN + p) * 32 + j] = (_Float16)xn;
        }
    }
    for (int m = 0; m < 4; ++m)
        ex16[ebase + (long)(cCIN - 1 + m) * cSEQ + t] = (_Float16)xm[((long)b * cSEQ + t) * 4 + m];
    if (t < cPN)
        for (int j = 16; j < 32; ++j) xt16[((long)b * cPN + t) * 32 + j] = (_Float16)0.f;
}

// ---------------- add positional encoding + global token ----------------
__global__ __launch_bounds__(256) void k_embed(const float* __restrict__ emb,
                                               const float* __restrict__ glb,
                                               float* __restrict__ x)
{
    int i = blockIdx.x * 256 + threadIdx.x;
    if (i >= cB * cL * cD) return;
    int d = i % cD; int r = i / cD; int l = r % cL; int b = r / cL;
    float v;
    if (l < cPN) {
        float div = expf(-9.2103403719761836f * (float)(d & ~1) / (float)cD);
        float a = (float)l * div;
        v = emb[((long)b * cPN + l) * cD + d] + ((d & 1) ? cosf(a) : sinf(a));
    } else v = glb[d];
    x[i] = v;
}

// ---------------- f32->f16 converts ----------------
__global__ __launch_bounds__(256) void k_cvt(const float* __restrict__ s,
                                             _Float16* __restrict__ d, int n)
{
    int i = blockIdx.x * 256 + threadIdx.x;
    if (i < n) d[i] = (_Float16)s[i];
}
__global__ __launch_bounds__(256) void k_cvt_pad(const float* __restrict__ s,
                                                 _Float16* __restrict__ d,
                                                 int N, int K, int Kp)
{
    int i = blockIdx.x * 256 + threadIdx.x;
    if (i >= N * Kp) return;
    int n = i / Kp, k = i - n * Kp;
    d[i] = (k < K) ? (_Float16)s[n * K + k] : (_Float16)0.f;
}

// ---------------- TCN gate: softmax over 2 logits ----------------
__global__ __launch_bounds__(256) void k_gate(const float* __restrict__ x,
                                              const float* __restrict__ gw,
                                              const float* __restrict__ gb,
                                              float* __restrict__ g3, float* __restrict__ g5)
{
    __shared__ float s0[256], s1[256];
    int b = blockIdx.x, t = threadIdx.x;
    float a0 = 0.f, a1 = 0.f;
    for (int d = t; d < cD; d += 256) {
        float s = 0.f;
        for (int l = 0; l < cL; ++l) s += x[((long)b * cL + l) * cD + d];
        s *= (1.0f / (float)cL);
        a0 += s * gw[d];
        a1 += s * gw[cD + d];
    }
    s0[t] = a0; s1[t] = a1; __syncthreads();
    for (int st = 128; st > 0; st >>= 1) { if (t < st) { s0[t] += s0[t + st]; s1[t] += s1[t + st]; } __syncthreads(); }
    if (t == 0) {
        float l0 = s0[0] + gb[0], l1 = s1[0] + gb[1];
        float m = fmaxf(l0, l1);
        float e0 = __expf(l0 - m), e1 = __expf(l1 - m);
        float inv = 1.f / (e0 + e1);
        g3[b] = e0 * inv; g5[b] = e1 * inv;
    }
}

// ---------------- im2col for conv1d (channels-last input) ----------------
__global__ __launch_bounds__(256) void k_im2col(const float* __restrict__ x,
                                                _Float16* __restrict__ out,
                                                int KW, int pad)
{
    int bl = blockIdx.x; int b = bl / cL, l = bl - b * cL;
    const float* xb = x + (long)b * cL * cD;
    _Float16* ob = out + (long)bl * cD * KW;
    for (int i = threadIdx.x; i < cD * KW; i += 256) {
        int din = i / KW, t = i - din * KW;
        int ls = l - pad + t;
        float v = (ls >= 0 && ls < cL) ? xb[(long)ls * cD + din] : 0.f;
        ob[i] = (_Float16)v;
    }
}

// ---------------- LayerNorm (D=512). mode 0: LN(y); 1: LN(res+y); 2: LN(y)+res
__global__ __launch_bounds__(128) void k_ln(const float* __restrict__ res, long rstr,
                                            const float* __restrict__ y, long ystr,
                                            const float* __restrict__ g,
                                            const float* __restrict__ bt,
                                            float* __restrict__ out, long ostr,
                                            _Float16* __restrict__ out16, long o16str,
                                            int mode)
{
    __shared__ float row[cD];
    __shared__ float sb[128];
    long r = blockIdx.x;
    const float* yr = y + r * ystr;
    const float* rr = res ? (res + r * rstr) : nullptr;
    float part = 0.f;
    for (int d = threadIdx.x; d < cD; d += 128) {
        float v = yr[d];
        if (mode == 1) v += rr[d];
        row[d] = v; part += v;
    }
    sb[threadIdx.x] = part; __syncthreads();
    for (int st = 64; st > 0; st >>= 1) { if (threadIdx.x < st) sb[threadIdx.x] += sb[threadIdx.x + st]; __syncthreads(); }
    float mean = sb[0] / (float)cD; __syncthreads();
    float p2 = 0.f;
    for (int d = threadIdx.x; d < cD; d += 128) { float t = row[d] - mean; p2 += t * t; }
    sb[threadIdx.x] = p2; __syncthreads();
    for (int st = 64; st > 0; st >>= 1) { if (threadIdx.x < st) sb[threadIdx.x] += sb[threadIdx.x + st]; __syncthreads(); }
    float inv = rsqrtf(sb[0] / (float)cD + 1e-5f);
    for (int d = threadIdx.x; d < cD; d += 128) {
        float v = (row[d] - mean) * inv * g[d] + bt[d];
        if (mode == 2) v += rr[d];
        out[r * ostr + d] = v;
        if (out16) out16[r * o16str + d] = (_Float16)v;
    }
}

// ---------------- DFT twiddle tables ----------------
__global__ __launch_bounds__(256) void k_dft(float* __restrict__ ct, float* __restrict__ st,
                                             int S, int bins)
{
    int i = blockIdx.x * 256 + threadIdx.x;
    if (i >= bins * S) return;
    int f = i / S, s = i - f * S;
    float a = -6.283185307179586f * (float)(f * s) / (float)S;
    ct[i] = cosf(a); st[i] = sinf(a);
}

// ---------------- Fourier magnitude top-k -> source mask ----------------
__global__ __launch_bounds__(128) void k_fourier_mask(
    const _Float16* __restrict__ Kmat, long sBatch,
    const float* __restrict__ ct, const float* __restrict__ st,
    unsigned char* __restrict__ mask, int S, int bins, int P, int kk)
{
    __shared__ _Float16 ktile[cL * cDK];
    __shared__ float mags[80];
    int z = blockIdx.x; int b = z >> 3, h = z & 7;
    const _Float16* Kb = Kmat + (long)b * sBatch + h * cDK;
    for (int i = threadIdx.x; i < S * cDK; i += 128) {
        int s = i >> 6, e = i & 63;
        ktile[i] = Kb[(long)s * cD + e];
    }
    __syncthreads();
    for (int f = threadIdx.x; f < bins; f += 128) {
        const float* cr = ct + f * S;
        const float* sr = st + f * S;
        float acc = 0.f;
        for (int e = 0; e < cDK; ++e) {
            float re = 0.f, im = 0.f;
            for (int s = 0; s < S; ++s) {
                float kv = (float)ktile[(s << 6) + e];
                re += kv * cr[s];
                im += kv * sr[s];
            }
            acc += sqrtf(re * re + im * im);
        }
        mags[f] = acc;
    }
    __syncthreads();
    unsigned char* mb = mask + (long)z * P;
    for (int i = threadIdx.x; i < P; i += 128) mb[i] = 0;
    __syncthreads();
    if (threadIdx.x == 0) {
        for (int t = 0; t < kk; ++t) {
            float best = -1.f; int bi = 0;
            for (int f = 0; f < bins; ++f) if (mags[f] > best) { best = mags[f]; bi = f; }
            mags[bi] = -2.f; mb[bi] = 1;
        }
    }
}

// ---------------- masked softmax -> f16 probs (zero-padded to P) ----------
__global__ __launch_bounds__(128) void k_softmax(const float* __restrict__ scores,
                                                 _Float16* __restrict__ attn,
                                                 const unsigned char* __restrict__ mask,
                                                 int S, int ld, int P, int Lq)
{
    __shared__ float sb[128];
    int l = blockIdx.x, z = blockIdx.y;
    const float* sr = scores + ((long)z * Lq + l) * ld;
    _Float16* ar = attn + ((long)z * Lq + l) * P;
    const unsigned char* mb = mask + (long)z * P;
    float mx = -1e30f;
    for (int s = threadIdx.x; s < S; s += 128)
        if (mb[s]) mx = fmaxf(mx, sr[s] * cSCALE);
    sb[threadIdx.x] = mx; __syncthreads();
    for (int st = 64; st > 0; st >>= 1) { if (threadIdx.x < st) sb[threadIdx.x] = fmaxf(sb[threadIdx.x], sb[threadIdx.x + st]); __syncthreads(); }
    mx = sb[0]; __syncthreads();
    float sum = 0.f;
    for (int s = threadIdx.x; s < S; s += 128)
        if (mb[s]) sum += __expf(sr[s] * cSCALE - mx);
    sb[threadIdx.x] = sum; __syncthreads();
    for (int st = 64; st > 0; st >>= 1) { if (threadIdx.x < st) sb[threadIdx.x] += sb[threadIdx.x + st]; __syncthreads(); }
    float inv = 1.f / sb[0];
    for (int s = threadIdx.x; s < P; s += 128) {
        float o = 0.f;
        if (s < S && mb[s]) o = __expf(sr[s] * cSCALE - mx) * inv;
        ar[s] = (_Float16)o;
    }
}

// ---------------- (B,L,D) f16 -> (B, D*L) flat for head ----------------
__global__ __launch_bounds__(256) void k_flat(const _Float16* __restrict__ x16,
                                              _Float16* __restrict__ flat)
{
    int i = blockIdx.x * 256 + threadIdx.x;
    if (i >= cB * cD * cL) return;
    int b = i / (cD * cL); int r = i - b * (cD * cL);
    int d = r / cL; int l = r - d * cL;
    flat[i] = x16[((long)(b * cL + l)) * cD + d];
}

// ---------------- de-normalize to output ----------------
__global__ __launch_bounds__(256) void k_denorm(const float* __restrict__ dec,
                                                const float* __restrict__ meanb,
                                                const float* __restrict__ stdb,
                                                float* __restrict__ out)
{
    int i = blockIdx.x * 256 + threadIdx.x;
    if (i >= cB * cPRED) return;
    int b = i / cPRED;
    out[i] = dec[i] * stdb[b * cCIN + cCIN - 1] + meanb[b * cCIN + cCIN - 1];
}

// =====================================================================
static void gemm(hipStream_t s,
                 const _Float16* A, int lda, long sAo, long sAi,
                 const _Float16* W, int ldw, long sWo, long sWi,
                 float* C, _Float16* C16, int ldc, long sCo, long sCi,
                 const float* bias, const float* gscale,
                 int M, int N, int K, int nbatch, int bdiv, int flags)
{
    int mt = (M + 15) / 16, nt64 = (N + 63) / 64;
    dim3 g(mt, (nt64 + 3) / 4, nbatch);
    k_gemm<<<g, 128, 0, s>>>(A, lda, sAo, sAi, W, ldw, sWo, sWi,
                             C, C16, ldc, sCo, sCi, bias, gscale,
                             M, N, K, bdiv, flags);
}

struct LP { const float *b1,*b2,*b3,*c1,*c2,
                        *ca_bk,*ca_bo,*ca_bq,*ca_bv,*ca_wk,*ca_wo,*ca_wq,*ca_wv,
                        *g1,*g2,*g3,
                        *sa_bk,*sa_bo,*sa_bq,*sa_bv,*sa_wk,*sa_wo,*sa_wq,*sa_wv,
                        *w1,*w2; };
struct LW { _Float16 *saq,*sak,*sav,*sao,*caq,*cak,*cav,*cao,*w1,*w2; };

extern "C" void kernel_launch(void* const* d_in, const int* in_sizes, int n_in,
                              void* d_out, int out_size, void* d_ws, size_t ws_size,
                              hipStream_t stream)
{
    (void)in_sizes; (void)n_in; (void)out_size; (void)ws_size;
    // ---- inputs (jax tree order: top-level dict insertion, params sorted recursively)
    const float* x_enc  = (const float*)d_in[0];
    const float* x_mark = (const float*)d_in[1];
    const float* p_en_w   = (const float*)d_in[4];
    const float* p_ex_b   = (const float*)d_in[5];
    const float* p_ex_w   = (const float*)d_in[6];
    const float* p_glb    = (const float*)d_in[7];
    const float* p_head_b = (const float*)d_in[8];
    const float* p_head_w = (const float*)d_in[9];
    LP lp[2];
    for (int li = 0; li < 2; ++li) {
        int base = 10 + li * 26;
        const float** q = (const float**)&lp[li];
        for (int k = 0; k < 26; ++k) q[k] = (const float*)d_in[base + k];
    }
    const float* p_norm_b = (const float*)d_in[62];
    const float* p_norm_g = (const float*)d_in[63];
    const float* p_t_b3   = (const float*)d_in[64];
    const float* p_t_b5   = (const float*)d_in[65];
    const float* p_gate_b = (const float*)d_in[66];
    const float* p_gate_w = (const float*)d_in[67];
    const float* p_t_lnb  = (const float*)d_in[68];
    const float* p_t_lng  = (const float*)d_in[69];
    const float* p_proj_b = (const float*)d_in[70];
    const float* p_proj_w = (const float*)d_in[71];
    const float* p_w3     = (const float*)d_in[72];
    const float* p_w5     = (const float*)d_in[73];

    // ---- workspace bump allocator
    char* p = (char*)d_ws;
    auto alloc = [&](size_t bytes) -> char* {
        char* r = p; p += (bytes + 255) & ~(size_t)255; return r;
    };
    float* meanb  = (float*)alloc((size_t)cB * cCIN * 4);
    float* stdb   = (float*)alloc((size_t)cB * cCIN * 4);
    float* x      = (float*)alloc((size_t)cB * cL * cD * 4);
    float* t0     = (float*)alloc((size_t)cB * cL * cD * 4);
    float* scores = (float*)alloc((size_t)cB * cH * cL * cSP * 4);
    float* g3buf  = (float*)alloc(cB * 4);
    float* g5buf  = (float*)alloc(cB * 4);
    float* decb   = (float*)alloc((size_t)cB * cPRED * 4);
    float* ctab   = (float*)alloc((size_t)cSBINS * cL * 4);
    float* stab   = (float*)alloc((size_t)cSBINS * cL * 4);
    float* ctabc  = (float*)alloc((size_t)cCBINS * cCS * 4);
    float* stabc  = (float*)alloc((size_t)cCBINS * cCS * 4);
    // arena aliasing: {im2col3,im2col5} (TCN) vs {ffn y1, attn probs} (encoder)
    char* arena = alloc((size_t)70 * 1024 * 1024);
    _Float16* im3    = (_Float16*)arena;
    _Float16* im5    = (_Float16*)(arena + (size_t)27 * 1024 * 1024);
    _Float16* y116   = (_Float16*)arena;
    _Float16* attn16 = (_Float16*)(arena + (size_t)35 * 1024 * 1024);
    _Float16* xt16    = (_Float16*)alloc((size_t)cB * cPN * 32 * 2);
    _Float16* ex16    = (_Float16*)alloc((size_t)cB * cCS * cSEQ * 2);
    _Float16* cross16 = (_Float16*)alloc((size_t)cB * cCS * cD * 2);
    _Float16* x16     = (_Float16*)alloc((size_t)cB * cL * cD * 2);
    _Float16* comb16  = (_Float16*)alloc((size_t)cB * cL * 1024 * 2);
    _Float16* q16     = (_Float16*)alloc((size_t)cB * cL * cD * 2);
    _Float16* k16     = (_Float16*)alloc((size_t)cB * cL * cD * 2);
    _Float16* vt16    = (_Float16*)alloc((size_t)cB * cD * cSP * 2);
    _Float16* ao16    = (_Float16*)alloc((size_t)cB * cL * cD * 2);
    _Float16* kc16    = (_Float16*)alloc((size_t)cB * cCS * cD * 2);
    _Float16* vtc16   = (_Float16*)alloc((size_t)cB * cD * cCP * 2);
    _Float16* attnc16 = (_Float16*)alloc((size_t)cB * cH * cCP * 2);
    _Float16* qg16    = (_Float16*)alloc((size_t)cB * cD * 2);
    _Float16* aoc16   = (_Float16*)alloc((size_t)cB * cD * 2);
    _Float16* flat16  = (_Float16*)alloc((size_t)cB * cD * cL * 2);
    unsigned char* maskS = (unsigned char*)alloc((size_t)cB * cH * cSP);
    unsigned char* maskC = (unsigned char*)alloc((size_t)cB * cH * cCP);
    // f16 weights
    _Float16* en16   = (_Float16*)alloc((size_t)cD * 32 * 2);
    _Float16* exw16  = (_Float16*)alloc((size_t)cD * cSEQ * 2);
    _Float16* proj16 = (_Float16*)alloc((size_t)cD * 1024 * 2);
    _Float16* w316   = (_Float16*)alloc((size_t)cD * cD * 3 * 2);
    _Float16* w516   = (_Float16*)alloc((size_t)cD * cD * 5 * 2);
    _Float16* head16 = (_Float16*)alloc((size_t)cPRED * cD * cL * 2);
    LW lw[2];
    for (int li = 0; li < 2; ++li) {
        _Float16** q = (_Float16**)&lw[li];
        for (int k = 0; k < 8; ++k) q[k] = (_Float16*)alloc((size_t)cD * cD * 2);
        lw[li].w1 = (_Float16*)alloc((size_t)cDFF * cD * 2);
        lw[li].w2 = (_Float16*)alloc((size_t)cD * cDFF * 2);
    }

    auto cvt = [&](const float* s, _Float16* d, int n) {
        k_cvt<<<(n + 255) / 256, 256, 0, stream>>>(s, d, n);
    };

    // zero-init padded transposed-V buffers (pad rows never written by GEMMs)
    hipMemsetAsync((void*)vt16, 0, (size_t)cB * cD * cSP * 2, stream);
    hipMemsetAsync((void*)vtc16, 0, (size_t)cB * cD * cCP * 2, stream);

    // ---- stage 0: stats, normalized inputs, weight conversion, DFT tables
    k_stats<<<cB * cCIN, 256, 0, stream>>>(x_enc, meanb, stdb);
    k_prep<<<(cB * cSEQ + 255) / 256, 256, 0, stream>>>(x_enc, x_mark, meanb, stdb, xt16, ex16);
    k_cvt_pad<<<(cD * 32 + 255) / 256, 256, 0, stream>>>(p_en_w, en16, cD, 16, 32);
    cvt(p_ex_w, exw16, cD * cSEQ);
    cvt(p_proj_w, proj16, cD * 1024);
    cvt(p_w3, w316, cD * cD * 3);
    cvt(p_w5, w516, cD * cD * 5);
    cvt(p_head_w, head16, cPRED * cD * cL);
    for (int li = 0; li < 2; ++li) {
        cvt(lp[li].sa_wq, lw[li].saq, cD * cD); cvt(lp[li].sa_wk, lw[li].sak, cD * cD);
        cvt(lp[li].sa_wv, lw[li].sav, cD * cD); cvt(lp[li].sa_wo, lw[li].sao, cD * cD);
        cvt(lp[li].ca_wq, lw[li].caq, cD * cD); cvt(lp[li].ca_wk, lw[li].cak, cD * cD);
        cvt(lp[li].ca_wv, lw[li].cav, cD * cD); cvt(lp[li].ca_wo, lw[li].cao, cD * cD);
        cvt(lp[li].w1, lw[li].w1, cDFF * cD);   cvt(lp[li].w2, lw[li].w2, cD * cDFF);
    }
    k_dft<<<(cSBINS * cL + 255) / 256, 256, 0, stream>>>(ctab, stab, cL, cSBINS);
    k_dft<<<(cCBINS * cCS + 255) / 256, 256, 0, stream>>>(ctabc, stabc, cCS, cCBINS);

    // ---- patch embedding (K padded 16->32) + PE + global token
    gemm(stream, xt16, 32, (long)cPN * 32, 0, en16, 32, 0, 0,
         t0, nullptr, cD, (long)cPN * cD, 0, nullptr, nullptr,
         cPN, cD, 32, cB, 1, 0);
    k_embed<<<(cB * cL * cD + 255) / 256, 256, 0, stream>>>(t0, p_glb, x);

    // ---- TCN: gated dual conv (im2col + WMMA GEMM), proj, LN(y)+x
    k_gate<<<cB, 256, 0, stream>>>(x, p_gate_w, p_gate_b, g3buf, g5buf);
    k_im2col<<<cB * cL, 256, 0, stream>>>(x, im3, 3, 1);
    k_im2col<<<cB * cL, 256, 0, stream>>>(x, im5, 5, 2);
    gemm(stream, im3, cD * 3, (long)cL * cD * 3, 0, w316, cD * 3, 0, 0,
         nullptr, comb16, 1024, (long)cL * 1024, 0, p_t_b3, g3buf,
         cL, cD, cD * 3, cB, 1, 1 | 2 | 8);
    gemm(stream, im5, cD * 5, (long)cL * cD * 5, 0, w516, cD * 5, 0, 0,
         nullptr, comb16 + cD, 1024, (long)cL * 1024, 0, p_t_b5, g5buf,
         cL, cD, cD * 5, cB, 1, 1 | 2 | 8);
    gemm(stream, comb16, 1024, (long)cL * 1024, 0, proj16, 1024, 0, 0,
         t0, nullptr, cD, (long)cL * cD, 0, p_proj_b, nullptr,
         cL, cD, 1024, cB, 1, 0);
    k_ln<<<cB * cL, 128, 0, stream>>>(x, cD, t0, cD, p_t_lng, p_t_lnb,
                                      x, cD, x16, cD, 2);

    // ---- cross features: (B,35,2048) @ ex_w^T
    gemm(stream, ex16, cSEQ, (long)cCS * cSEQ, 0, exw16, cSEQ, 0, 0,
         nullptr, cross16, cD, (long)cCS * cD, 0, p_ex_b, nullptr,
         cCS, cD, cSEQ, cB, 1, 2);

    // ---- encoder layers
    for (int li = 0; li < 2; ++li) {
        const LP& L = lp[li]; const LW& Wt = lw[li];
        // self-attention projections
        gemm(stream, x16, cD, (long)cL * cD, 0, Wt.saq, cD, 0, 0,
             nullptr, q16, cD, (long)cL * cD, 0, L.sa_bq, nullptr, cL, cD, cD, cB, 1, 2);
        gemm(stream, x16, cD, (long)cL * cD, 0, Wt.sak, cD, 0, 0,
             nullptr, k16, cD, (long)cL * cD, 0, L.sa_bk, nullptr, cL, cD, cD, cB, 1, 2);
        gemm(stream, x16, cD, (long)cL * cD, 0, Wt.sav, cD, 0, 0,
             nullptr, vt16, cSP, (long)cD * cSP, 0, L.sa_bv, nullptr, cL, cD, cD, cB, 1, 2 | 4);
        // Fourier top-k source mask from K
        k_fourier_mask<<<cB * cH, 128, 0, stream>>>(k16, (long)cL * cD, ctab, stab,
                                                    maskS, cL, cSBINS, cSP, 10);
        // scores = Q @ K^T per (b,h)
        gemm(stream, q16, cD, (long)cL * cD, cDK, k16, cD, (long)cL * cD, cDK,
             scores, nullptr, cSP, (long)cH * cL * cSP, (long)cL * cSP,
             nullptr, nullptr, cL, cL, cDK, cB * cH, cH, 0);
        k_softmax<<<dim3(cL, cB * cH), 128, 0, stream>>>(scores, attn16, maskS,
                                                         cL, cSP, cSP, cL);
        // attn @ V (V pre-transposed, K padded to 160 with zeros)
        gemm(stream, attn16, cSP, (long)cH * cL * cSP, (long)cL * cSP,
             vt16, cSP, (long)cD * cSP, (long)cDK * cSP,
             nullptr, ao16, cD, (long)cL * cD, cDK,
             nullptr, nullptr, cL, cDK, cSP, cB * cH, cH, 2);
        gemm(stream, ao16, cD, (long)cL * cD, 0, Wt.sao, cD, 0, 0,
             t0, nullptr, cD, (long)cL * cD, 0, L.sa_bo, nullptr, cL, cD, cD, cB, 1, 0);
        k_ln<<<cB * cL, 128, 0, stream>>>(x, cD, t0, cD, L.g1, L.b1, x, cD, x16, cD, 1);

        // cross-attention on last token
        gemm(stream, x16 + (long)cPN * cD, cD, (long)cL * cD, 0, Wt.caq, cD, 0, 0,
             nullptr, qg16, cD, cD, 0, L.ca_bq, nullptr, 1, cD, cD, cB, 1, 2);
        gemm(stream, cross16, cD, (long)cCS * cD, 0, Wt.cak, cD, 0, 0,
             nullptr, kc16, cD, (long)cCS * cD, 0, L.ca_bk, nullptr, cCS, cD, cD, cB, 1, 2);
        gemm(stream, cross16, cD, (long)cCS * cD, 0, Wt.cav, cD, 0, 0,
             nullptr, vtc16, cCP, (long)cD * cCP, 0, L.ca_bv, nullptr, cCS, cD, cD, cB, 1, 2 | 4);
        k_fourier_mask<<<cB * cH, 128, 0, stream>>>(kc16, (long)cCS * cD, ctabc, stabc,
                                                    maskC, cCS, cCBINS, cCP, 10);
        gemm(stream, qg16, cD, cD, cDK, kc16, cD, (long)cCS * cD, cDK,
             scores, nullptr, cCP, (long)cH * cCP, cCP,
             nullptr, nullptr, 1, cCS, cDK, cB * cH, cH, 0);
        k_softmax<<<dim3(1, cB * cH), 128, 0, stream>>>(scores, attnc16, maskC,
                                                        cCS, cCP, cCP, 1);
        gemm(stream, attnc16, cCP, (long)cH * cCP, cCP,
             vtc16, cCP, (long)cD * cCP, (long)cDK * cCP,
             nullptr, aoc16, cD, cD, cDK,
             nullptr, nullptr, 1, cDK, cCP, cB * cH, cH, 2);
        gemm(stream, aoc16, cD, cD, 0, Wt.cao, cD, 0, 0,
             t0, nullptr, cD, cD, 0, L.ca_bo, nullptr, 1, cD, cD, cB, 1, 0);
        k_ln<<<cB, 128, 0, stream>>>(x + (long)cPN * cD, (long)cL * cD, t0, cD, L.g2, L.b2,
                                     x + (long)cPN * cD, (long)cL * cD,
                                     x16 + (long)cPN * cD, (long)cL * cD, 1);

        // FFN
        gemm(stream, x16, cD, (long)cL * cD, 0, Wt.w1, cD, 0, 0,
             nullptr, y116, cDFF, (long)cL * cDFF, 0, L.c1, nullptr, cL, cDFF, cD, cB, 1, 1 | 2);
        gemm(stream, y116, cDFF, (long)cL * cDFF, 0, Wt.w2, cDFF, 0, 0,
             t0, nullptr, cD, (long)cL * cD, 0, L.c2, nullptr, cL, cD, cDFF, cB, 1, 0);
        k_ln<<<cB * cL, 128, 0, stream>>>(x, cD, t0, cD, L.g3, L.b3, x, cD, x16, cD, 1);
    }

    // ---- final norm, head, de-normalize
    k_ln<<<cB * cL, 128, 0, stream>>>(nullptr, 0, x, cD, p_norm_g, p_norm_b,
                                      x, cD, x16, cD, 0);
    k_flat<<<(cB * cD * cL + 255) / 256, 256, 0, stream>>>(x16, flat16);
    gemm(stream, flat16, cD * cL, (long)cD * cL, 0, head16, cD * cL, 0, 0,
         decb, nullptr, cPRED, cPRED, 0, p_head_b, nullptr,
         1, cPRED, cD * cL, cB, 1, 0);
    k_denorm<<<(cB * cPRED + 255) / 256, 256, 0, stream>>>(decb, meanb, stdb, (float*)d_out);
}
